// GridNeRF_2207613190673
// MI455X (gfx1250) — compile-verified, matrix operand order checked by reading, not run on hardware
//
#include <hip/hip_runtime.h>
#include <math.h>

typedef __attribute__((ext_vector_type(16))) _Float16 v16h;
typedef __attribute__((ext_vector_type(8)))  _Float16 v8h;
typedef __attribute__((ext_vector_type(8)))  float    v8f;

#define KP     96          // padded K stride (halves) for activation/W1/W3 tiles
#define TSIZE  (1u << 19)  // hash table entries per level
#define NWAVE  4           // waves per block; 16 points per wave-tile
#define MAXBLK 1024        // grid-stride: each wave loops over ~8 tiles

// Load a 16x32 f16 WMMA fragment (A or B operand) from an LDS tile.
// Tile layout: row-major [row][k], row stride = `stride` halves (16B aligned).
// CDNA5 16-bit operand layout: lanes 0-15 hold K{0..7,16..23}, lanes 16-31 K{8..15,24..31}.
__device__ __forceinline__ v16h ld_frag(const _Float16* base, int stride, int row,
                                        int kbase, int khalf) {
  const _Float16* p = base + row * stride + kbase + khalf * 8;
  v16h out;
  ((v8h*)&out)[0] = *(const v8h*)(p);
  ((v8h*)&out)[1] = *(const v8h*)(p + 16);
  return out;
}

__global__ __launch_bounds__(128) void nerf_fused(
    const float* __restrict__ x, const float* __restrict__ v,
    const float* __restrict__ table,
    const float* __restrict__ dw1, const float* __restrict__ db1,
    const float* __restrict__ dw2, const float* __restrict__ db2,
    const float* __restrict__ rw1, const float* __restrict__ rb1,
    const float* __restrict__ rw2, const float* __restrict__ rb2,
    float* __restrict__ sigma_out, float* __restrict__ rgb_out, int npts)
{
  // Weights, transposed to [n][k] f16 so B fragments are two b128 LDS reads.
  __shared__ _Float16 W1t[64 * KP];   // dw1: 71x64  -> [64][96]
  __shared__ _Float16 W2t[80 * 64];   // dw2: 64x65  -> [80][64]
  __shared__ _Float16 W3t[64 * KP];   // rw1: 73x64  -> [64][96]
  __shared__ _Float16 W4t[16 * 64];   // rw2: 64x3   -> [16][64]
  __shared__ _Float16 bufE[NWAVE][16 * KP];  // encoding / layer-3 input
  __shared__ _Float16 bufH[NWAVE][16 * KP];  // hidden activations

  const int tid = threadIdx.x;

  // ---- stage weights once per block (amortized over the tile loop) ----
  for (int i = tid; i < 64 * KP; i += 128) {
    int nn = i / KP, kk = i % KP;
    W1t[i] = (kk < 71) ? (_Float16)dw1[kk * 64 + nn] : (_Float16)0.f;
    W3t[i] = (kk < 73) ? (_Float16)rw1[kk * 64 + nn] : (_Float16)0.f;
  }
  for (int i = tid; i < 80 * 64; i += 128) {
    int nn = i / 64, kk = i % 64;
    W2t[i] = (nn < 65) ? (_Float16)dw2[kk * 65 + nn] : (_Float16)0.f;
  }
  for (int i = tid; i < 16 * 64; i += 128) {
    int nn = i / 64, kk = i % 64;
    W4t[i] = (nn < 3) ? (_Float16)rw2[kk * 3 + nn] : (_Float16)0.f;
  }
  __syncthreads();

  const int wave  = tid >> 5;
  const int lane  = tid & 31;
  const int nlo   = lane & 15;     // column / point-row selector
  const int khalf = lane >> 4;     // K-half selector for fragments

  _Float16* enc = bufE[wave];
  _Float16* hid = bufH[wave];

  // ---- hoist loop-invariant per-lane biases into registers ----
  float b1v[4], b2v[5], b3v[4];
  #pragma unroll
  for (int nt = 0; nt < 4; ++nt) b1v[nt] = db1[nt * 16 + nlo];
  #pragma unroll
  for (int nt = 0; nt < 5; ++nt) {
    int col = nt * 16 + nlo;
    b2v[nt] = (col < 65) ? db2[col] : 0.f;
  }
  #pragma unroll
  for (int nt = 0; nt < 4; ++nt) b3v[nt] = rb1[nt * 16 + nlo];
  const float b4v = rb2[nlo < 3 ? nlo : 0];

  const int tilestep = gridDim.x * NWAVE;

  for (int tile = blockIdx.x * NWAVE + wave; (tile + 1) * 16 <= npts; tile += tilestep) {
    const int base = tile * 16;

    // prefetch next tile's coordinates while we compute this one
    {
      long nb = (long)(tile + tilestep) * 16;
      __builtin_prefetch(x + nb * 3, 0, 3);
      __builtin_prefetch(v + nb * 3, 0, 3);
    }

    // =========================== encoding ===========================
    // enc dims: [0..2]=xyz, [3..38]=fourier, [39..46]=grid lv0..3 (mask kills
    // levels 4..15 => dims 47..70 are zero), [71..95]=pad zeros.
    {
      const int p = nlo;  // each lane does scalar work for one of the 16 points
      const float xx = x[(base + p) * 3 + 0];
      const float xy = x[(base + p) * 3 + 1];
      const float xz = x[(base + p) * 3 + 2];
      if (khalf == 0) {
        // xyz + fourier + zero-fill tail
        enc[p * KP + 0] = (_Float16)xx;
        enc[p * KP + 1] = (_Float16)xy;
        enc[p * KP + 2] = (_Float16)xz;
        float cval[3] = {xx, xy, xz};
        #pragma unroll
        for (int c = 0; c < 3; ++c) {
          float f = 1.f;
          #pragma unroll
          for (int j = 0; j < 6; ++j) {
            float s, co;
            __sincosf(cval[c] * f, &s, &co);
            enc[p * KP + 3 + c * 12 + j]     = (_Float16)s;
            enc[p * KP + 3 + c * 12 + 6 + j] = (_Float16)co;
            f *= 2.f;
          }
        }
        enc[p * KP + 47] = (_Float16)0.f;
        v8h z8 = {};
        #pragma unroll
        for (int i = 48; i < KP; i += 8) *(v8h*)(enc + p * KP + i) = z8;
      } else {
        // hash-grid levels 0..3 (levels >=4 are masked to zero upstream)
        const float RESF[4] = {33.f, 43.f, 56.f, 74.f};
        const float ux = (xx + 1.f) * 0.5f;
        const float uy = (xy + 1.f) * 0.5f;
        const float uz = (xz + 1.f) * 0.5f;
        const float2* tab2 = (const float2*)table;
        #pragma unroll
        for (int lv = 0; lv < 4; ++lv) {
          float px = ux * RESF[lv], py = uy * RESF[lv], pz = uz * RESF[lv];
          float fx = floorf(px), fy = floorf(py), fz = floorf(pz);
          float tx = px - fx, ty = py - fy, tz = pz - fz;
          float wx = tx * tx * (3.f - 2.f * tx);
          float wy = ty * ty * (3.f - 2.f * ty);
          float wz = tz * tz * (3.f - 2.f * tz);
          unsigned cx = (unsigned)fx, cy = (unsigned)fy, cz = (unsigned)fz;
          float f0 = 0.f, f1 = 0.f;
          #pragma unroll
          for (int c = 0; c < 8; ++c) {
            unsigned ox = c & 1, oy = (c >> 1) & 1, oz = (c >> 2) & 1;
            unsigned h = ((cx + ox) ^ ((cy + oy) * 2654435761u)
                                    ^ ((cz + oz) * 805459861u)) & (TSIZE - 1u);
            float2 tv = tab2[(size_t)lv * TSIZE + h];
            float cw = (ox ? wx : 1.f - wx) * (oy ? wy : 1.f - wy)
                     * (oz ? wz : 1.f - wz);
            f0 += tv.x * cw;
            f1 += tv.y * cw;
          }
          enc[p * KP + 39 + lv * 2 + 0] = (_Float16)f0;
          enc[p * KP + 39 + lv * 2 + 1] = (_Float16)f1;
        }
      }
    }

    // ================= layer 1: enc(71,pad96) -> 64, ReLU =================
    {
      v8f acc[4] = {};
      #pragma unroll
      for (int kc = 0; kc < 3; ++kc) {
        v16h a = ld_frag(enc, KP, nlo, kc * 32, khalf);
        #pragma unroll
        for (int nt = 0; nt < 4; ++nt) {
          v16h b = ld_frag(W1t, KP, nt * 16 + nlo, kc * 32, khalf);
          acc[nt] = __builtin_amdgcn_wmma_f32_16x16x32_f16(
              false, a, false, b, (short)0, acc[nt], false, false);
        }
      }
      #pragma unroll
      for (int nt = 0; nt < 4; ++nt) {
        int col = nt * 16 + nlo;
        #pragma unroll
        for (int r = 0; r < 8; ++r) {
          int m = r + khalf * 8;
          hid[m * KP + col] = (_Float16)fmaxf(acc[nt][r] + b1v[nt], 0.f);
        }
      }
    }

    // ========== layer 2: hid(64) -> 65 ; sigma=exp(col0), feat=ReLU ==========
    {
      v8f acc[5] = {};
      #pragma unroll
      for (int kc = 0; kc < 2; ++kc) {
        v16h a = ld_frag(hid, KP, nlo, kc * 32, khalf);
        #pragma unroll
        for (int nt = 0; nt < 5; ++nt) {
          v16h b = ld_frag(W2t, 64, nt * 16 + nlo, kc * 32, khalf);
          acc[nt] = __builtin_amdgcn_wmma_f32_16x16x32_f16(
              false, a, false, b, (short)0, acc[nt], false, false);
        }
      }
      #pragma unroll
      for (int nt = 0; nt < 5; ++nt) {
        int col = nt * 16 + nlo;
        #pragma unroll
        for (int r = 0; r < 8; ++r) {
          int m = r + khalf * 8;
          float val = acc[nt][r] + b2v[nt];
          if (col == 0)       sigma_out[base + m] = __expf(val);
          else if (col < 65)  enc[m * KP + (col - 1)] = (_Float16)fmaxf(val, 0.f);
        }
      }
      // append sh3(v) at dims 64..72, zero 73..95 (layer-3 input K=73 pad 96)
      if (khalf == 0) {
        const int p = nlo;
        float vx = v[(base + p) * 3 + 0];
        float vy = v[(base + p) * 3 + 1];
        float vz = v[(base + p) * 3 + 2];
        enc[p * KP + 64] = (_Float16)0.28209479177387814f;
        enc[p * KP + 65] = (_Float16)(-0.4886025119029199f * vy);
        enc[p * KP + 66] = (_Float16)( 0.4886025119029199f * vz);
        enc[p * KP + 67] = (_Float16)(-0.4886025119029199f * vx);
        enc[p * KP + 68] = (_Float16)( 1.0925484305920792f * vx * vy);
        enc[p * KP + 69] = (_Float16)(-1.0925484305920792f * vy * vz);
        enc[p * KP + 70] = (_Float16)( 0.9461746957575601f * vz * vz - 0.31539156525252005f);
        enc[p * KP + 71] = (_Float16)(-1.0925484305920792f * vx * vz);
        enc[p * KP + 72] = (_Float16)( 0.5462742152960396f * (vx * vx - vy * vy));
        for (int i = 73; i < KP; ++i) enc[p * KP + i] = (_Float16)0.f;
      }
    }

    // ============== layer 3: [feat|sh3](73,pad96) -> 64, ReLU ==============
    {
      v8f acc[4] = {};
      #pragma unroll
      for (int kc = 0; kc < 3; ++kc) {
        v16h a = ld_frag(enc, KP, nlo, kc * 32, khalf);
        #pragma unroll
        for (int nt = 0; nt < 4; ++nt) {
          v16h b = ld_frag(W3t, KP, nt * 16 + nlo, kc * 32, khalf);
          acc[nt] = __builtin_amdgcn_wmma_f32_16x16x32_f16(
              false, a, false, b, (short)0, acc[nt], false, false);
        }
      }
      #pragma unroll
      for (int nt = 0; nt < 4; ++nt) {
        int col = nt * 16 + nlo;
        #pragma unroll
        for (int r = 0; r < 8; ++r) {
          int m = r + khalf * 8;
          hid[m * KP + col] = (_Float16)fmaxf(acc[nt][r] + b3v[nt], 0.f);
        }
      }
    }

    // ================ layer 4: hid(64) -> 3, sigmoid ================
    {
      v8f acc = {};
      #pragma unroll
      for (int kc = 0; kc < 2; ++kc) {
        v16h a = ld_frag(hid, KP, nlo, kc * 32, khalf);
        v16h b = ld_frag(W4t, 64, nlo, kc * 32, khalf);
        acc = __builtin_amdgcn_wmma_f32_16x16x32_f16(
            false, a, false, b, (short)0, acc, false, false);
      }
      if (nlo < 3) {
        #pragma unroll
        for (int r = 0; r < 8; ++r) {
          int m = r + khalf * 8;
          float val = acc[r] + b4v;
          rgb_out[(base + m) * 3 + nlo] = 1.f / (1.f + __expf(-val));
        }
      }
    }
  }
}

extern "C" void kernel_launch(void* const* d_in, const int* in_sizes, int n_in,
                              void* d_out, int out_size, void* d_ws, size_t ws_size,
                              hipStream_t stream) {
  const float* x     = (const float*)d_in[0];
  const float* v     = (const float*)d_in[1];
  // d_in[2] = o, unused by the forward pass
  const float* table = (const float*)d_in[3];
  const float* dw1   = (const float*)d_in[4];
  const float* db1   = (const float*)d_in[5];
  const float* dw2   = (const float*)d_in[6];
  const float* db2   = (const float*)d_in[7];
  const float* rw1   = (const float*)d_in[8];
  const float* rb1   = (const float*)d_in[9];
  const float* rw2   = (const float*)d_in[10];
  const float* rb2   = (const float*)d_in[11];

  const int n = in_sizes[0] / 3;           // 524288
  float* sigma = (float*)d_out;            // [N]
  float* rgb   = sigma + n;                // [N,3]

  const int pts_per_block = NWAVE * 16;    // 64
  int blocks = (n + pts_per_block - 1) / pts_per_block;
  if (blocks > MAXBLK) blocks = MAXBLK;    // grid-stride loop covers the rest
  if (blocks < 1) blocks = 1;
  nerf_fused<<<blocks, 128, 0, stream>>>(x, v, table, dw1, db1, dw2, db2,
                                         rw1, rb1, rw2, rb2, sigma, rgb, n);
}
